// Mamba3LanguageModel_55061480734891
// MI455X (gfx1250) — matrix operand order repaired and checked by reading, not compiled, not run on hardware
//
#include <hip/hip_runtime.h>
#include <hip/hip_bf16.h>

typedef unsigned short u16;
typedef unsigned int   u32;

typedef __attribute__((ext_vector_type(16))) __bf16 v16bf;
typedef __attribute__((ext_vector_type(8)))  __bf16 v8bf;
typedef __attribute__((ext_vector_type(8)))  float  v8f;

#define T_TOK   2048      // B*L = 2*1024
#define LSEQ    1024
#define DMODEL  768
#define VVOC    50257
#define VPAD    50304     // 128*393
#define DI_     1536
#define DXBC_   1792
#define HHEADS  24
#define NSTATE  128
#define PHEAD   64
#define DIN_    3352
#define DINPAD  3456      // 128*27
#define EPS_    1e-6f

// ---------------------------------------------------------------- helpers
__device__ __forceinline__ u16 f2bf(float f) {
    u32 u = __float_as_uint(f);
    u32 r = (u + 0x7FFFu + ((u >> 16) & 1u)) >> 16;
    return (u16)r;
}
__device__ __forceinline__ float siluf(float x) {
    return x / (1.0f + __expf(-x));
}

// sum-reduce across a 256-thread block (wave32 shuffles + 8-slot LDS)
__device__ __forceinline__ float blockSum256(float v, float* sh) {
    for (int o = 16; o > 0; o >>= 1) v += __shfl_xor(v, o, 32);
    if ((threadIdx.x & 31) == 0) sh[threadIdx.x >> 5] = v;
    __syncthreads();
    if (threadIdx.x == 0) {
        float a = 0.f;
        #pragma unroll
        for (int i = 0; i < 8; ++i) a += sh[i];
        sh[0] = a;
    }
    __syncthreads();
    float r = sh[0];
    __syncthreads();
    return r;
}

// ---------------------------------------------------------------- async-to-LDS staging (gfx1250)
#define HAVE_ASYNC_LDS __has_builtin(__builtin_amdgcn_global_load_async_to_lds_b128)

#if HAVE_ASYNC_LDS
// builtin's param 0 is 'int __attribute__((vector_size(16))) *' (per clang diagnostic)
typedef int vi4 __attribute__((vector_size(16)));
typedef __attribute__((address_space(3))) vi4 as3_vi4;
// copy 32 bytes (16 bf16) from global to LDS via two async b128 transfers;
// instruction IOFFSET applies to both global and LDS addresses (ISA 10.x)
__device__ __forceinline__ void stage32_async(const u16* g, u16* l) {
    __builtin_amdgcn_global_load_async_to_lds_b128((vi4*)(void*)g, (as3_vi4*)l, 0, 0);
    __builtin_amdgcn_global_load_async_to_lds_b128((vi4*)(void*)g, (as3_vi4*)l, 16, 0);
}
__device__ __forceinline__ void wait_async0() {
#if __has_builtin(__builtin_amdgcn_s_wait_asynccnt)
    __builtin_amdgcn_s_wait_asynccnt(0);
#else
    asm volatile("s_wait_asynccnt 0x0" ::: "memory");
#endif
}
#endif

// ---------------------------------------------------------------- weight prep
// W is K x N row-major; out Bt is Npad x K row-major bf16 (zero-padded rows)
__global__ __launch_bounds__(256) void k_transpose_bf16(
    const float* __restrict__ W, u16* __restrict__ Bt, int K, int N, int Npad) {
    int i = blockIdx.x * 256 + threadIdx.x;
    if (i >= Npad * K) return;
    int n = i / K, k = i - n * K;
    float v = (n < N) ? W[(size_t)k * N + n] : 0.f;
    Bt[i] = f2bf(v);
}
// W is N x K row-major already; convert + pad rows to Npad
__global__ __launch_bounds__(256) void k_convert_pad_bf16(
    const float* __restrict__ W, u16* __restrict__ Bt, int K, int N, int Npad) {
    int i = blockIdx.x * 256 + threadIdx.x;
    if (i >= Npad * K) return;
    int n = i / K, k = i - n * K;
    float v = (n < N) ? W[(size_t)n * K + k] : 0.f;
    Bt[i] = f2bf(v);
}

// ---------------------------------------------------------------- embedding
__global__ __launch_bounds__(256) void k_embed(
    const int* __restrict__ idx, const float* __restrict__ emb, float* __restrict__ x) {
    int i = blockIdx.x * 256 + threadIdx.x;   // over T_TOK*DMODEL
    int t = i / DMODEL, d = i - t * DMODEL;
    x[i] = emb[(size_t)idx[t] * DMODEL + d];
}

// ---------------------------------------------------------------- rmsnorm -> bf16
__global__ __launch_bounds__(256) void k_rmsnorm_bf16(
    const float* __restrict__ x, const float* __restrict__ w,
    u16* __restrict__ out, int D) {
    __shared__ float sh[8];
    int t = blockIdx.x, tid = threadIdx.x;
    const float* xr = x + (size_t)t * D;
    float vals[8];
    float ss = 0.f;
    int i = 0;
    for (int d = tid; d < D; d += 256, ++i) { float v = xr[d]; vals[i] = v; ss += v * v; }
    float tot = blockSum256(ss, sh);
    float rs = rsqrtf(tot / (float)D + EPS_);
    i = 0;
    for (int d = tid; d < D; d += 256, ++i)
        out[(size_t)t * D + d] = f2bf(vals[i] * rs * w[d]);
}

// ---------------------------------------------------------------- WMMA fragment loaders
// LDS tile: 128 rows x 32 K, row-major u16(bf16)
__device__ __forceinline__ v16bf load_frag_A(const u16* lds, int rowBase, int lane) {
    int half = lane >> 4;                 // ISA 16-bit A layout: chunks of 8 K
    const u16* p = lds + (rowBase + (lane & 15)) * 32;
    v8bf lo = *(const v8bf*)(p + half * 8);
    v8bf hi = *(const v8bf*)(p + 16 + half * 8);
    return __builtin_shufflevector(lo, hi, 0,1,2,3,4,5,6,7,8,9,10,11,12,13,14,15);
}
__device__ __forceinline__ v16bf load_frag_B(const u16* lds, int rowBase, int lane) {
    int half = lane >> 4;                 // ISA B layout: chunks of 8 K (same as A)
    const u16* p = lds + (rowBase + (lane & 15)) * 32;
    v8bf lo = *(const v8bf*)(p + half * 8);
    v8bf hi = *(const v8bf*)(p + 16 + half * 8);
    return __builtin_shufflevector(lo, hi, 0,1,2,3,4,5,6,7,8,9,10,11,12,13,14,15);
}

// ---------------------------------------------------------------- NT GEMM: C[MxN] = A[MxK] * Bt[NxK]^T (+Cadd)
// grid = (Npad/128, M/128), block = 256 (8 waves: 2 along M x 4 along N)
// Double-buffered LDS pipeline; global->LDS via async-to-LDS when available.
__global__ __launch_bounds__(256) void k_gemm_nt(
    const u16* __restrict__ A, const u16* __restrict__ Bt,
    float* __restrict__ C, const float* __restrict__ Cadd,
    int Nreal, int K) {
    __shared__ __align__(16) u16 As[2][128 * 32];
    __shared__ __align__(16) u16 Bs[2][128 * 32];

    const int tid   = threadIdx.x;
    const int lane  = tid & 31;
    const int wave  = tid >> 5;
    const int waveM = wave >> 2;          // 0..1  -> 64 rows each
    const int waveN = wave & 3;           // 0..3  -> 32 cols each
    const int blockM = blockIdx.y * 128;
    const int blockN = blockIdx.x * 128;
    const int lrow  = tid >> 1;           // 0..127
    const int lhalf = tid & 1;            // 0..1
    const int loff  = lrow * 32 + lhalf * 16;

    v8f acc[4][2] = {};

    const u16* aSrc = A  + (size_t)(blockM + lrow) * K + lhalf * 16;
    const u16* bSrc = Bt + (size_t)(blockN + lrow) * K + lhalf * 16;

    auto compute_tile = [&](const u16* Asb, const u16* Bsb) {
        v16bf bfr[2];
        #pragma unroll
        for (int nt = 0; nt < 2; ++nt)
            bfr[nt] = load_frag_B(Bsb, waveN * 32 + nt * 16, lane);
        #pragma unroll
        for (int mt = 0; mt < 4; ++mt) {
            v16bf afr = load_frag_A(Asb, waveM * 64 + mt * 16, lane);
            #pragma unroll
            for (int nt = 0; nt < 2; ++nt)
                acc[mt][nt] = __builtin_amdgcn_wmma_f32_16x16x32_bf16(
                    false, afr, false, bfr[nt], (short)0, acc[mt][nt], false, false);
        }
    };

#if HAVE_ASYNC_LDS
    // ---- async-to-LDS double-buffered pipeline ----
    stage32_async(aSrc, &As[0][loff]);
    stage32_async(bSrc, &Bs[0][loff]);
    for (int k0 = 0; k0 < K; k0 += 32) {
        const int cur = (k0 >> 5) & 1;
        wait_async0();                    // this wave's copies into buf[cur] done
        __syncthreads();                  // all waves done (and done reading buf[cur^1])
        if (k0 + 32 < K) {                // kick off next tile while we compute
            stage32_async(aSrc + k0 + 32, &As[cur ^ 1][loff]);
            stage32_async(bSrc + k0 + 32, &Bs[cur ^ 1][loff]);
        }
        compute_tile(As[cur], Bs[cur]);
    }
#else
    // ---- register-staged double-buffered pipeline ----
    uint4 ra0, ra1, rb0, rb1;
    { const uint4* s = (const uint4*)(aSrc); ra0 = s[0]; ra1 = s[1]; }
    { const uint4* s = (const uint4*)(bSrc); rb0 = s[0]; rb1 = s[1]; }
    for (int k0 = 0; k0 < K; k0 += 32) {
        const int cur = (k0 >> 5) & 1;
        { uint4* d = (uint4*)(&As[cur][loff]); d[0] = ra0; d[1] = ra1; }
        { uint4* d = (uint4*)(&Bs[cur][loff]); d[0] = rb0; d[1] = rb1; }
        __syncthreads();
        if (k0 + 32 < K) {                // loads for next tile overlap this tile's WMMAs
            const uint4* s0 = (const uint4*)(aSrc + k0 + 32); ra0 = s0[0]; ra1 = s0[1];
            const uint4* s1 = (const uint4*)(bSrc + k0 + 32); rb0 = s1[0]; rb1 = s1[1];
            if (k0 + 64 < K) {
                __builtin_prefetch(aSrc + k0 + 64, 0, 1);
                __builtin_prefetch(bSrc + k0 + 64, 0, 1);
            }
        }
        compute_tile(As[cur], Bs[cur]);
    }
#endif

    // D layout: VGPR r -> (M = r + 8*half, N = lane&15)
    #pragma unroll
    for (int mt = 0; mt < 4; ++mt) {
        #pragma unroll
        for (int nt = 0; nt < 2; ++nt) {
            int col = blockN + waveN * 32 + nt * 16 + (lane & 15);
            if (col >= Nreal) continue;
            int row0 = blockM + waveM * 64 + mt * 16 + ((lane >> 4) << 3);
            #pragma unroll
            for (int r = 0; r < 8; ++r) {
                size_t o = (size_t)(row0 + r) * Nreal + col;
                float v = acc[mt][nt][r];
                if (Cadd) v += Cadd[o];
                C[o] = v;
            }
        }
    }
}

// ---------------------------------------------------------------- causal depthwise conv + SiLU
__global__ __launch_bounds__(256) void k_conv_silu(
    const float* __restrict__ zx, const float* __restrict__ cw,
    const float* __restrict__ cb, float* __restrict__ xbc) {
    int i = blockIdx.x * 256 + threadIdx.x;   // over T_TOK*DXBC_
    int tt = i / DXBC_, c = i - tt * DXBC_;
    int b = tt >> 10, l = tt & 1023;
    float acc = cb[c];
    #pragma unroll
    for (int j = 0; j < 4; ++j) {
        int ls = l - 3 + j;
        if (ls >= 0)
            acc += cw[c * 4 + j] * zx[(size_t)(b * LSEQ + ls) * DIN_ + DI_ + c];
    }
    xbc[i] = siluf(acc);
}

// ---------------------------------------------------------------- dt softplus + decay
__global__ __launch_bounds__(256) void k_dtdecay(
    const float* __restrict__ zx, const float* __restrict__ dt_bias,
    const float* __restrict__ A_log, float* __restrict__ dt, float* __restrict__ dec) {
    int i = blockIdx.x * 256 + threadIdx.x;   // over T_TOK*HHEADS
    if (i >= T_TOK * HHEADS) return;
    int tt = i / HHEADS, h = i - tt * HHEADS;
    float raw = zx[(size_t)tt * DIN_ + (DI_ + DXBC_) + h] + dt_bias[h];
    float d = (raw > 20.f) ? raw : log1pf(__expf(raw));
    float A = -__expf(A_log[h]);
    dt[i]  = d;
    dec[i] = __expf(d * A);
}

// ---------------------------------------------------------------- SSM scan
// grid = 48 (b*H), block = 256: thread = (p = tid>>2, g = tid&3); state 32 f32/thread
__global__ __launch_bounds__(256) void k_scan(
    const float* __restrict__ xbc, const float* __restrict__ dtv,
    const float* __restrict__ dec, float* __restrict__ y) {
    __shared__ float Bsh[32 * 128], Csh[32 * 128], Xsh[32 * 64], Dd[32], Dt[32];
    int blk = blockIdx.x;
    int b = blk / HHEADS, h = blk - b * HHEADS;
    int tid = threadIdx.x;
    int p = tid >> 2, g = tid & 3, nb = g * 32;
    int tbase = b * LSEQ;

    float hs[32];
    #pragma unroll
    for (int j = 0; j < 32; ++j) hs[j] = 0.f;

    for (int ch = 0; ch < LSEQ / 32; ++ch) {
        int t0 = ch * 32;
        __syncthreads();
        for (int i = tid; i < 32 * 128; i += 256) {
            int s = i >> 7, n = i & 127;
            size_t row = (size_t)(tbase + t0 + s) * DXBC_;
            Bsh[i] = xbc[row + DI_ + n];
            Csh[i] = xbc[row + DI_ + NSTATE + n];
        }
        for (int i = tid; i < 32 * 64; i += 256) {
            int s = i >> 6, pp = i & 63;
            Xsh[i] = xbc[(size_t)(tbase + t0 + s) * DXBC_ + h * PHEAD + pp];
        }
        if (tid < 32) {
            Dd[tid] = dec[(size_t)(tbase + t0 + tid) * HHEADS + h];
            Dt[tid] = dtv[(size_t)(tbase + t0 + tid) * HHEADS + h];
        }
        __syncthreads();
        for (int s = 0; s < 32; ++s) {
            float d = Dd[s];
            float a = Dt[s] * Xsh[s * 64 + p];
            const float* Bp = Bsh + s * 128 + nb;
            const float* Cp = Csh + s * 128 + nb;
            float part = 0.f;
            #pragma unroll
            for (int j = 0; j < 32; ++j) {
                hs[j] = d * hs[j] + a * Bp[j];
                part += hs[j] * Cp[j];
            }
            part += __shfl_xor(part, 1, 32);
            part += __shfl_xor(part, 2, 32);
            if (g == 0)
                y[(size_t)(tbase + t0 + s) * DI_ + h * PHEAD + p] = part;
        }
    }
}

// ---------------------------------------------------------------- gate: (y + Dp*x) * silu(z), rmsnorm, -> bf16
__global__ __launch_bounds__(256) void k_gate(
    const float* __restrict__ y, const float* __restrict__ xbc,
    const float* __restrict__ zx, const float* __restrict__ Dp,
    const float* __restrict__ normw, u16* __restrict__ out) {
    __shared__ float sh[8];
    int t = blockIdx.x, tid = threadIdx.x;
    float vals[8];
    float ss = 0.f;
    int i = 0;
    for (int di = tid; di < DI_; di += 256, ++i) {
        int h = di >> 6;
        float yv = y[(size_t)t * DI_ + di] + Dp[h] * xbc[(size_t)t * DXBC_ + di];
        float zv = zx[(size_t)t * DIN_ + di];
        float v = yv * siluf(zv);
        vals[i] = v;
        ss += v * v;
    }
    float tot = blockSum256(ss, sh);
    float rs = rsqrtf(tot / (float)DI_ + EPS_);
    i = 0;
    for (int di = tid; di < DI_; di += 256, ++i)
        out[(size_t)t * DI_ + di] = f2bf(vals[i] * rs * normw[di]);
}

// ---------------------------------------------------------------- loss
__global__ void k_zero(float* p) { *p = 0.f; }

__global__ __launch_bounds__(256) void k_loss(
    const float* __restrict__ logits, const int* __restrict__ targets,
    float* __restrict__ loss) {
    __shared__ float sh[8];
    int t = blockIdx.x, tid = threadIdx.x;
    const float* row = logits + (size_t)t * VVOC;
    float m = -3.4e38f;
    for (int v = tid; v < VVOC; v += 256) m = fmaxf(m, row[v]);
    for (int o = 16; o > 0; o >>= 1) m = fmaxf(m, __shfl_xor(m, o, 32));
    if ((tid & 31) == 0) sh[tid >> 5] = m;
    __syncthreads();
    if (tid == 0) {
        float mm = sh[0];
        for (int i = 1; i < 8; ++i) mm = fmaxf(mm, sh[i]);
        sh[0] = mm;
    }
    __syncthreads();
    m = sh[0];
    __syncthreads();
    float s = 0.f;
    for (int v = tid; v < VVOC; v += 256) s += __expf(row[v] - m);
    float tot = blockSum256(s, sh);
    if (tid == 0) {
        float lp = row[targets[t]] - m - logf(tot);
        atomicAdd(loss, -lp * (1.0f / (float)T_TOK));
    }
}

// ---------------------------------------------------------------- host
extern "C" void kernel_launch(void* const* d_in, const int* in_sizes, int n_in,
                              void* d_out, int out_size, void* d_ws, size_t ws_size,
                              hipStream_t stream) {
    (void)in_sizes; (void)n_in; (void)out_size; (void)ws_size;

    const int*   idx     = (const int*)d_in[0];
    const int*   targets = (const int*)d_in[1];
    const float* emb     = (const float*)d_in[2];
    const float* norm_f  = (const float*)d_in[3 + 12 * 9];

    // workspace bump allocator (256B aligned)
    char* ws = (char*)d_ws;
    size_t off = 0;
    auto alloc = [&](size_t bytes) -> char* {
        off = (off + 255) & ~(size_t)255;
        char* p = ws + off;
        off += bytes;
        return p;
    };

    u16* embT = (u16*)alloc((size_t)VPAD * DMODEL * sizeof(u16));
    u16* winT[12]; u16* woutT[12];
    for (int i = 0; i < 12; ++i) {
        winT[i]  = (u16*)alloc((size_t)DINPAD * DMODEL * sizeof(u16));
        woutT[i] = (u16*)alloc((size_t)DMODEL * DI_ * sizeof(u16));
    }
    float* x    = (float*)alloc((size_t)T_TOK * DMODEL * sizeof(float));
    u16*   xnbf = (u16*)  alloc((size_t)T_TOK * DMODEL * sizeof(u16));
    float* zx   = (float*)alloc((size_t)T_TOK * DIN_   * sizeof(float));
    float* xbc  = (float*)alloc((size_t)T_TOK * DXBC_  * sizeof(float));
    float* dtb  = (float*)alloc((size_t)T_TOK * HHEADS * sizeof(float));
    float* dcb  = (float*)alloc((size_t)T_TOK * HHEADS * sizeof(float));
    float* yb   = (float*)alloc((size_t)T_TOK * DI_    * sizeof(float));
    u16*   gat  = (u16*)  alloc((size_t)T_TOK * DI_    * sizeof(u16));

    // --- weight preprocessing (bf16, transposed to NxK, zero-padded) ---
    {
        int tot = VPAD * DMODEL;
        k_convert_pad_bf16<<<(tot + 255) / 256, 256, 0, stream>>>(emb, embT, DMODEL, VVOC, VPAD);
    }
    for (int i = 0; i < 12; ++i) {
        const float* w_in  = (const float*)d_in[3 + i * 9 + 1];
        const float* w_out = (const float*)d_in[3 + i * 9 + 8];
        int t1 = DINPAD * DMODEL;
        k_transpose_bf16<<<(t1 + 255) / 256, 256, 0, stream>>>(w_in, winT[i], DMODEL, DIN_, DINPAD);
        int t2 = DMODEL * DI_;
        k_transpose_bf16<<<(t2 + 255) / 256, 256, 0, stream>>>(w_out, woutT[i], DI_, DMODEL, DMODEL);
    }

    // --- embedding gather ---
    k_embed<<<(T_TOK * DMODEL) / 256, 256, 0, stream>>>(idx, emb, x);

    // --- layers ---
    for (int i = 0; i < 12; ++i) {
        int base = 3 + i * 9;
        const float* rms_w   = (const float*)d_in[base + 0];
        const float* conv_w  = (const float*)d_in[base + 2];
        const float* conv_b  = (const float*)d_in[base + 3];
        const float* dt_bias = (const float*)d_in[base + 4];
        const float* A_log   = (const float*)d_in[base + 5];
        const float* Dp      = (const float*)d_in[base + 6];
        const float* norm_w  = (const float*)d_in[base + 7];

        k_rmsnorm_bf16<<<T_TOK, 256, 0, stream>>>(x, rms_w, xnbf, DMODEL);
        k_gemm_nt<<<dim3(DINPAD / 128, T_TOK / 128), 256, 0, stream>>>(
            xnbf, winT[i], zx, nullptr, DIN_, DMODEL);
        k_conv_silu<<<(T_TOK * DXBC_) / 256, 256, 0, stream>>>(zx, conv_w, conv_b, xbc);
        k_dtdecay<<<(T_TOK * HHEADS + 255) / 256, 256, 0, stream>>>(zx, dt_bias, A_log, dtb, dcb);
        k_scan<<<2 * HHEADS, 256, 0, stream>>>(xbc, dtb, dcb, yb);
        k_gate<<<T_TOK, 256, 0, stream>>>(yb, xbc, zx, Dp, norm_w, gat);
        k_gemm_nt<<<dim3(DMODEL / 128, T_TOK / 128), 256, 0, stream>>>(
            gat, woutT[i], x, x, DMODEL, DI_);   // + residual
    }

    // --- final norm + logits + loss ---
    float* logits = (float*)d_out;
    float* loss   = logits + (size_t)T_TOK * VVOC;

    k_rmsnorm_bf16<<<T_TOK, 256, 0, stream>>>(x, norm_f, xnbf, DMODEL);
    k_gemm_nt<<<dim3(VPAD / 128, T_TOK / 128), 256, 0, stream>>>(
        xnbf, embT, logits, nullptr, VVOC, DMODEL);
    k_zero<<<1, 1, 0, stream>>>(loss);
    k_loss<<<T_TOK, 256, 0, stream>>>(logits, targets, loss);
}
